// MultiHeadAttention_44006234915100
// MI455X (gfx1250) — matrix operand order repaired and checked by reading, not compiled, not run on hardware
//
#include <hip/hip_runtime.h>
#include <hip/hip_bf16.h>

// ---------------------------------------------------------------------------
// Fused MHA forward for MI455X (gfx1250, wave32, WMMA + TDM).
//   B=2, S=2048, H=16, Dk=Dv=64, Dmodel=1024.
//  - All GEMMs: v_wmma_f32_16x16x32_f16, double-buffered LDS tiles filled by
//    the Tensor Data Mover (tensor_load_to_lds, TENSORcnt), HW LDS padding.
//  - Attention: two-pass online softmax; the 512MB attn tensor is written
//    exactly once; K/V re-reads hit the 192MB L2.
// ---------------------------------------------------------------------------

typedef __attribute__((ext_vector_type(16))) _Float16 v16h;
typedef __attribute__((ext_vector_type(8)))  _Float16 v8h;
typedef __attribute__((ext_vector_type(8)))  float    v8f;
typedef __attribute__((ext_vector_type(4)))  float    v4f;
typedef __attribute__((ext_vector_type(4)))  unsigned int v4u;
typedef __attribute__((ext_vector_type(8)))  int      v8i;
typedef __attribute__((ext_vector_type(4)))  int      v4i;

#define S_LEN   2048
#define NHEAD   16
#define DHEAD   64
#define DMODEL  1024
#define NROWS   4096          // B*S
#define LDP     40            // padded LDS row stride in halves (TDM pad 16dw+4dw)

static __device__ __forceinline__ v16h pack16(v8h lo, v8h hi) {
  v16h r;
#pragma unroll
  for (int i = 0; i < 8; ++i) { r[i] = lo[i]; r[8 + i] = hi[i]; }
  return r;
}

static __device__ __forceinline__ v8f wmma_f16(v16h a, v16h b, v8f c) {
  return __builtin_amdgcn_wmma_f32_16x16x32_f16(
      false, a, false, b, (short)0, c, false, false);
}

// ---------------------------------------------------------------------------
// TDM: async 2-D f16 tile load (tile_x contiguous elements per line, tile_y
// lines) from global to LDS, with HW padding of 4 DWORDs after every 16
// DWORDs stored -> LDS row stride = 32 + 8 = 40 halves (LDP).
// Descriptor packing per CDNA5 ISA ch.8 (group0 128b, group1 256b).
// This toolchain exposes the 6-arg builtin:
//   (v4u g0, v8i g1, v4i g2, v4i g3, v8i extra, i32 cpol)
// ---------------------------------------------------------------------------
static __device__ __forceinline__ void tdm_load_tile_f16(
    const _Float16* gsrc, unsigned lds_byte_addr,
    unsigned tile_x, unsigned tile_y,
    unsigned tensor_w, unsigned tensor_h, unsigned stride_elems) {
  const unsigned long long ga = (unsigned long long)(size_t)gsrc;
  v4u g0;
  g0[0] = 1u;                                       // count=1, user descriptor
  g0[1] = lds_byte_addr;                            // lds_addr [63:32]
  g0[2] = (unsigned)(ga & 0xffffffffu);             // global_addr[31:0]
  g0[3] = (unsigned)((ga >> 32) & 0x01ffffffu)      // global_addr[56:32]
        | (2u << 30);                               // type=2 ("image")
  v8i g1;
  g1[0] = (int)((1u << 16)                          // data_size=1 -> 2 bytes
              | (1u << 20)                          // pad_enable
              | (3u << 22)                          // pad_interval: 16 DWORDs
              | (3u << 25));                        // pad_amount:   4 DWORDs
  g1[1] = (int)((tensor_w & 0xffffu) << 16);        // tensor_dim0[15:0] @ [63:48]
  g1[2] = (int)(((tensor_w >> 16) & 0xffffu)        // tensor_dim0[31:16]
              | ((tensor_h & 0xffffu) << 16));      // tensor_dim1[15:0]
  g1[3] = (int)(((tensor_h >> 16) & 0xffffu)        // tensor_dim1[31:16]
              | ((tile_x & 0xffffu) << 16));        // tile_dim0
  g1[4] = (int)(tile_y & 0xffffu);                  // tile_dim1 (tile_dim2=0)
  g1[5] = (int)stride_elems;                        // tensor_dim0_stride[31:0]
  g1[6] = 0;                                        // stride[47:32], dim1_stride lo
  g1[7] = 0;
  const v4i z4 = {0, 0, 0, 0};
  const v8i z8 = {0, 0, 0, 0, 0, 0, 0, 0};
  __builtin_amdgcn_tensor_load_to_lds(g0, g1, z4, z4, z8, 0);
}

// ---------------------------------------------------------------------------
// Prep kernels: f32 -> f16 convert, and f32 -> f16 transpose (weights).
// ---------------------------------------------------------------------------
__global__ __launch_bounds__(256) void cvt_f16(const float* __restrict__ src,
                                               _Float16* __restrict__ dst) {
  const size_t i = ((size_t)blockIdx.x * 256 + threadIdx.x) * 4;
  v4f f = *(const v4f*)(src + i);
#pragma unroll
  for (int e = 0; e < 4; ++e) dst[i + e] = (_Float16)f[e];
}

__global__ __launch_bounds__(256) void transpose_cvt_f16(
    const float* __restrict__ W, _Float16* __restrict__ Wt) {
  // W[k][n] (1024x1024 f32) -> Wt[n][k] f16, 64x64 tile per block
  __shared__ _Float16 t[64 * 72];
  const int k0 = blockIdx.y * 64, n0 = blockIdx.x * 64;
  const int r = threadIdx.x >> 2, cb = (threadIdx.x & 3) * 16;
  {
    const float* src = W + (size_t)(k0 + r) * DMODEL + n0 + cb;
#pragma unroll
    for (int j = 0; j < 16; ++j) t[(cb + j) * 72 + r] = (_Float16)src[j];
  }
  __syncthreads();
  {
    _Float16* dst = Wt + (size_t)(n0 + r) * DMODEL + k0 + cb;
    const _Float16* s = t + r * 72 + cb;
#pragma unroll
    for (int j = 0; j < 16; ++j) dst[j] = s[j];
  }
}

// ---------------------------------------------------------------------------
// Shared GEMM core: 128x64 block tile, 8 waves (4x2), k-step 32,
// double-buffered TDM staging. A: f16 [M][1024]; B: f16 transposed [N][1024].
// ---------------------------------------------------------------------------
#define GEMM_TDM_PIPELINE(A_PTR, B_PTR)                                        \
  const int tid = threadIdx.x, lane = tid & 31, w = tid >> 5;                  \
  const int wr = w & 3, wc = w >> 2;                                           \
  const int m0 = blockIdx.y * 128, n0 = blockIdx.x * 64;                       \
  __shared__ _Float16 lA[2 * 128 * LDP];                                       \
  __shared__ _Float16 lB[2 * 64 * LDP];                                        \
  v8f acc[2][2] = {};                                                          \
  if (w == 0) {                                                                \
    tdm_load_tile_f16(A_PTR + (size_t)m0 * DMODEL, (unsigned)(size_t)&lA[0],   \
                      32, 128, DMODEL, NROWS, DMODEL);                         \
    tdm_load_tile_f16(B_PTR + (size_t)n0 * DMODEL, (unsigned)(size_t)&lB[0],   \
                      32, 64, DMODEL, DMODEL, DMODEL);                         \
    __builtin_amdgcn_s_wait_tensorcnt(0);                                      \
  }                                                                            \
  __syncthreads();                                                             \
  int cur = 0;                                                                 \
  for (int k0 = 0; k0 < DMODEL; k0 += 32) {                                    \
    if (w == 0 && k0 + 32 < DMODEL) {                                          \
      const int nxt = cur ^ 1;                                                 \
      tdm_load_tile_f16(A_PTR + (size_t)m0 * DMODEL + k0 + 32,                 \
                        (unsigned)(size_t)&lA[nxt * 128 * LDP],                \
                        32, 128, DMODEL, NROWS, DMODEL);                       \
      tdm_load_tile_f16(B_PTR + (size_t)n0 * DMODEL + k0 + 32,                 \
                        (unsigned)(size_t)&lB[nxt * 64 * LDP],                 \
                        32, 64, DMODEL, DMODEL, DMODEL);                       \
    }                                                                          \
    const _Float16* la = lA + cur * 128 * LDP;                                 \
    const _Float16* lb = lB + cur * 64 * LDP;                                  \
    v16h afrag[2], bfrag[2];                                                   \
    const int kbA = (lane >> 4) * 16;                                          \
    _Pragma("unroll")                                                          \
    for (int i = 0; i < 2; ++i) {                                              \
      const _Float16* p = la + (wr * 32 + i * 16 + (lane & 15)) * LDP;         \
      afrag[i] = pack16(*(const v8h*)(p + kbA), *(const v8h*)(p + kbA + 8));   \
    }                                                                          \
    const int kbB = (lane >> 4) * 16;                                          \
    _Pragma("unroll")                                                          \
    for (int j = 0; j < 2; ++j) {                                              \
      const _Float16* p = lb + (wc * 32 + j * 16 + (lane & 15)) * LDP + kbB;   \
      bfrag[j] = pack16(*(const v8h*)(p), *(const v8h*)(p + 8));               \
    }                                                                          \
    _Pragma("unroll")                                                          \
    for (int i = 0; i < 2; ++i)                                                \
      _Pragma("unroll")                                                        \
      for (int j = 0; j < 2; ++j)                                              \
        acc[i][j] = wmma_f16(afrag[i], bfrag[j], acc[i][j]);                   \
    if (w == 0) __builtin_amdgcn_s_wait_tensorcnt(0);                          \
    __syncthreads();                                                           \
    cur ^= 1;                                                                  \
  }

// MODE 0: write f16 [b][h][s][d]; MODE 1: write f16 [b][h][d][s] (V^T)
template <int MODE>
__global__ __launch_bounds__(256) void qkv_gemm_tdm(
    const _Float16* __restrict__ A, const _Float16* __restrict__ Wt,
    _Float16* __restrict__ Out) {
  GEMM_TDM_PIPELINE(A, Wt)
#pragma unroll
  for (int i = 0; i < 2; ++i)
#pragma unroll
    for (int j = 0; j < 2; ++j)
#pragma unroll
      for (int r = 0; r < 8; ++r) {
        const int m = m0 + wr * 32 + i * 16 + r + 8 * (lane >> 4);
        const int n = n0 + wc * 32 + j * 16 + (lane & 15);
        const int b = m >> 11, s = m & (S_LEN - 1);
        const int h = n >> 6,  d = n & (DHEAD - 1);
        const size_t off = (MODE == 0)
            ? ((size_t)((b * NHEAD + h) * S_LEN + s) * DHEAD + d)
            : ((size_t)((b * NHEAD + h) * DHEAD + d) * S_LEN + s);
        Out[off] = (_Float16)acc[i][j][r];
      }
}

__global__ __launch_bounds__(256) void out_proj_gemm_tdm(
    const _Float16* __restrict__ A, const _Float16* __restrict__ Wt,
    const float* __restrict__ resid, float* __restrict__ out) {
  GEMM_TDM_PIPELINE(A, Wt)
#pragma unroll
  for (int i = 0; i < 2; ++i)
#pragma unroll
    for (int j = 0; j < 2; ++j)
#pragma unroll
      for (int r = 0; r < 8; ++r) {
        const int m = m0 + wr * 32 + i * 16 + r + 8 * (lane >> 4);
        const int n = n0 + wc * 32 + j * 16 + (lane & 15);
        const size_t off = (size_t)m * DMODEL + n;
        out[off] = acc[i][j][r] + resid[off];
      }
}

// ---------------------------------------------------------------------------
// Attention: grid = (S/128, B*H). 8 waves/block, 16 queries/wave.
// Pass 1: online max/sum (WMMA scores + wave32 shfl reductions).
// Pass 2: recompute scores, write normalized probs once, accumulate O = P@V.
// ---------------------------------------------------------------------------
__global__ __launch_bounds__(256) void attn_kernel(
    const _Float16* __restrict__ Qh, const _Float16* __restrict__ Kh,
    const _Float16* __restrict__ Vt, const unsigned char* __restrict__ mask,
    float* __restrict__ attn, _Float16* __restrict__ Ows) {
  const int tid = threadIdx.x, lane = tid & 31, w = tid >> 5;
  const int bh = blockIdx.y, b = bh >> 4, h = bh & 15;
  const int q0 = blockIdx.x * 128 + w * 16;

  const _Float16* Qb = Qh + (size_t)bh * S_LEN * DHEAD;
  const _Float16* Kb = Kh + (size_t)bh * S_LEN * DHEAD;
  const _Float16* Vb = Vt + (size_t)bh * DHEAD * S_LEN;
  float* attnb = attn + (size_t)bh * S_LEN * S_LEN;
  const unsigned char* mb = mask + (size_t)b * S_LEN * S_LEN;

  v16h aq[2];
  {
    const _Float16* qp = Qb + (size_t)(q0 + (lane & 15)) * DHEAD + ((lane >> 4) * 16);
#pragma unroll
    for (int c = 0; c < 2; ++c)
      aq[c] = pack16(*(const v8h*)(qp + 32 * c), *(const v8h*)(qp + 32 * c + 8));
  }

  const float invscale = 0.35355339059f;   // 1 / 64^0.25 (matches reference)
  float mrow[8], lrow[8];
#pragma unroll
  for (int r = 0; r < 8; ++r) { mrow[r] = -3.0e38f; lrow[r] = 0.f; }

  for (int kt = 0; kt < S_LEN; kt += 16) {
    const _Float16* kp = Kb + (size_t)(kt + (lane & 15)) * DHEAD + ((lane >> 4) * 16);
    if (kt + 16 < S_LEN) __builtin_prefetch(kp + 16 * DHEAD, 0, 1);
    v16h bk0 = pack16(*(const v8h*)(kp),      *(const v8h*)(kp + 8));
    v16h bk1 = pack16(*(const v8h*)(kp + 32), *(const v8h*)(kp + 40));
    v8f c = {};
    c = wmma_f16(aq[0], bk0, c);
    c = wmma_f16(aq[1], bk1, c);
    const int key = kt + (lane & 15);
#pragma unroll
    for (int r = 0; r < 8; ++r) {
      const int q = q0 + r + 8 * (lane >> 4);
      float s = c[r] * invscale;
      if (mb[(size_t)q * S_LEN + key]) s = -1.0e9f;
      float tmax = s;
#pragma unroll
      for (int o = 1; o < 16; o <<= 1) tmax = fmaxf(tmax, __shfl_xor(tmax, o, 16));
      const float nm = fmaxf(mrow[r], tmax);
      float p = __expf(s - nm);
#pragma unroll
      for (int o = 1; o < 16; o <<= 1) p += __shfl_xor(p, o, 16);
      lrow[r] = lrow[r] * __expf(mrow[r] - nm) + p;
      mrow[r] = nm;
    }
  }
  float rinv[8];
#pragma unroll
  for (int r = 0; r < 8; ++r) rinv[r] = 1.0f / lrow[r];

  __shared__ _Float16 lP[8 * 16 * LDP];
  _Float16* lPw = lP + w * 16 * LDP;

  v8f ov[4] = {};
  for (int kt = 0; kt < S_LEN; kt += 32) {
#pragma unroll
    for (int u = 0; u < 2; ++u) {
      const int k16 = kt + 16 * u;
      const _Float16* kp =
          Kb + (size_t)(k16 + (lane & 15)) * DHEAD + ((lane >> 4) * 16);
      v16h bk0 = pack16(*(const v8h*)(kp),      *(const v8h*)(kp + 8));
      v16h bk1 = pack16(*(const v8h*)(kp + 32), *(const v8h*)(kp + 40));
      v8f c = {};
      c = wmma_f16(aq[0], bk0, c);
      c = wmma_f16(aq[1], bk1, c);
      const int key = k16 + (lane & 15);
#pragma unroll
      for (int r = 0; r < 8; ++r) {
        const int q = q0 + r + 8 * (lane >> 4);
        float s = c[r] * invscale;
        if (mb[(size_t)q * S_LEN + key]) s = -1.0e9f;
        const float p = __expf(s - mrow[r]) * rinv[r];
        attnb[(size_t)q * S_LEN + key] = p;                     // written once
        lPw[(r + 8 * (lane >> 4)) * LDP + 16 * u + (lane & 15)] = (_Float16)p;
      }
    }
    __syncthreads();

    const int kbA = (lane >> 4) * 16;
    const _Float16* pp = lPw + (lane & 15) * LDP;
    v16h ap = pack16(*(const v8h*)(pp + kbA), *(const v8h*)(pp + kbA + 8));
#pragma unroll
    for (int t = 0; t < 4; ++t) {
      const _Float16* vp =
          Vb + (size_t)(16 * t + (lane & 15)) * S_LEN + kt + ((lane >> 4) * 16);
      v16h bv = pack16(*(const v8h*)(vp), *(const v8h*)(vp + 8));
      ov[t] = wmma_f16(ap, bv, ov[t]);
    }
    __syncthreads();
  }

#pragma unroll
  for (int t = 0; t < 4; ++t)
#pragma unroll
    for (int r = 0; r < 8; ++r) {
      const int q = q0 + r + 8 * (lane >> 4);
      const int d = 16 * t + (lane & 15);
      Ows[(size_t)(b * S_LEN + q) * DMODEL + h * DHEAD + d] = (_Float16)ov[t][r];
    }
}

// ---------------------------------------------------------------------------
// LayerNorm in place over each 1024-wide row.
// ---------------------------------------------------------------------------
__global__ __launch_bounds__(256) void ln_kernel(
    float* __restrict__ out, const float* __restrict__ gamma,
    const float* __restrict__ beta) {
  const int g = blockIdx.x, tid = threadIdx.x;
  float* row = out + (size_t)g * DMODEL;
  float x[4], s = 0.f, s2 = 0.f;
#pragma unroll
  for (int j = 0; j < 4; ++j) {
    x[j] = row[tid + 256 * j];
    s += x[j];
    s2 += x[j] * x[j];
  }
#pragma unroll
  for (int o = 1; o < 32; o <<= 1) {
    s  += __shfl_xor(s, o, 32);
    s2 += __shfl_xor(s2, o, 32);
  }
  __shared__ float rs[8], rs2[8], stats[2];
  if ((tid & 31) == 0) { rs[tid >> 5] = s; rs2[tid >> 5] = s2; }
  __syncthreads();
  if (tid == 0) {
    float a = 0.f, b2 = 0.f;
#pragma unroll
    for (int i = 0; i < 8; ++i) { a += rs[i]; b2 += rs2[i]; }
    const float mu = a * (1.0f / DMODEL);
    const float var = b2 * (1.0f / DMODEL) - mu * mu;
    stats[0] = mu;
    stats[1] = rsqrtf(var + 1e-5f);
  }
  __syncthreads();
  const float mu = stats[0], rstd = stats[1];
#pragma unroll
  for (int j = 0; j < 4; ++j) {
    const int c = tid + 256 * j;
    row[c] = (x[j] - mu) * rstd * gamma[c] + beta[c];
  }
}

// ---------------------------------------------------------------------------
extern "C" void kernel_launch(void* const* d_in, const int* in_sizes, int n_in,
                              void* d_out, int out_size, void* d_ws, size_t ws_size,
                              hipStream_t stream) {
  (void)in_sizes; (void)n_in; (void)out_size; (void)ws_size;
  const float* Qin   = (const float*)d_in[0];
  const float* Kin   = (const float*)d_in[1];
  const float* Vin   = (const float*)d_in[2];
  const unsigned char* maskp = (const unsigned char*)d_in[3];
  const float* Wq    = (const float*)d_in[4];
  const float* Wk    = (const float*)d_in[5];
  const float* Wv    = (const float*)d_in[6];
  const float* Wo    = (const float*)d_in[7];
  const float* gamma = (const float*)d_in[8];
  const float* beta  = (const float*)d_in[9];

  float* outBase  = (float*)d_out;                     // [B,S,DMODEL] f32
  float* attnBase = outBase + (size_t)NROWS * DMODEL;  // [B,H,S,S] f32

  const size_t NE  = (size_t)NROWS * DMODEL;           // 4,194,304 elements
  const size_t WE  = (size_t)DMODEL * DMODEL;          // 1,048,576 elements
  _Float16* Qh  = (_Float16*)d_ws;      // [b][h][s][d]
  _Float16* Kh  = Qh  + NE;             // [b][h][s][d]
  _Float16* Vt  = Kh  + NE;             // [b][h][d][s]
  _Float16* Ows = Vt  + NE;             // [b][s][h*64+d]
  _Float16* Xq  = Ows + NE;             // f16 copies of the inputs
  _Float16* Xk  = Xq  + NE;
  _Float16* Xv  = Xk  + NE;
  _Float16* Wtq = Xv  + NE;             // f16 transposed weights [n][k]
  _Float16* Wtk = Wtq + WE;
  _Float16* Wtv = Wtk + WE;
  _Float16* Wto = Wtv + WE;

  // --- prep: f16 copies + transposed f16 weights (one-time, ~100MB traffic) --
  cvt_f16<<<NE / 1024, 256, 0, stream>>>(Qin, Xq);
  cvt_f16<<<NE / 1024, 256, 0, stream>>>(Kin, Xk);
  cvt_f16<<<NE / 1024, 256, 0, stream>>>(Vin, Xv);
  dim3 gtr(DMODEL / 64, DMODEL / 64);
  transpose_cvt_f16<<<gtr, 256, 0, stream>>>(Wq, Wtq);
  transpose_cvt_f16<<<gtr, 256, 0, stream>>>(Wk, Wtk);
  transpose_cvt_f16<<<gtr, 256, 0, stream>>>(Wv, Wtv);
  transpose_cvt_f16<<<gtr, 256, 0, stream>>>(Wo, Wto);

  // --- projections: TDM-fed double-buffered WMMA GEMMs ---------------------
  dim3 gproj(DMODEL / 64, NROWS / 128);                // (16, 32)
  qkv_gemm_tdm<0><<<gproj, 256, 0, stream>>>(Xq, Wtq, Qh);
  qkv_gemm_tdm<0><<<gproj, 256, 0, stream>>>(Xk, Wtk, Kh);
  qkv_gemm_tdm<1><<<gproj, 256, 0, stream>>>(Xv, Wtv, Vt);

  // --- attention -----------------------------------------------------------
  dim3 gattn(S_LEN / 128, 2 * NHEAD);                  // (16, 32)
  attn_kernel<<<gattn, 256, 0, stream>>>(Qh, Kh, Vt, maskp, attnBase, Ows);

  // --- output projection + residual, then LayerNorm ------------------------
  out_proj_gemm_tdm<<<gproj, 256, 0, stream>>>(Ows, Wto, Qin, outBase);
  ln_kernel<<<NROWS, 256, 0, stream>>>(outBase, gamma, beta);
}
